// SimpleMultiheadAttention_72318659330189
// MI455X (gfx1250) — compile-verified, matrix operand order checked by reading, not run on hardware
//
#include <hip/hip_runtime.h>
#include <hip/hip_bf16.h>

#define D_IN    1024
#define D_MODEL 1024
#define NH      16
#define DH      64
#define BB      2
#define SS      2048
#define SCALE   0.125f   // 1/sqrt(64)

typedef __attribute__((ext_vector_type(16))) _Float16 v16h;
typedef __attribute__((ext_vector_type(8)))  float    v8f;
typedef _Float16 h16;

#define NEG_INF (-__builtin_inff())

// Output modes for the WMMA GEMM (Y = X @ W^T + bias)
#define OUT_F32_PLAIN 0   // f32 [M,N]
#define OUT_F16_PLAIN 1   // f16 [M,N]
#define OUT_F16_QK    2   // f16 [B,H,S,DH]  (m = b*S+s, n = h*DH+d)
#define OUT_F16_VT    3   // f16 [B,H,DH,S]

// ---------------------------------------------------------------------------
// Load a 16x32 f16 fragment (A layout; B uses the same per-lane K pattern with
// row = n-tile row of the [N,K] matrix).
// Lane L: element e -> k = (e>>3)*16 + hi8 + ((e>>1)&3)*2 + (e&1), hi8 = L>=16?8:0
// Contiguous pairs -> compiler emits two global_load_b128 per fragment.
// ---------------------------------------------------------------------------
__device__ __forceinline__ v16h load_frag(const h16* __restrict__ base, int ld,
                                          int row, int k0, int hi8) {
  const h16* p = base + (size_t)row * ld + k0 + hi8;
  v16h f;
#pragma unroll
  for (int hf = 0; hf < 2; ++hf)
#pragma unroll
    for (int i = 0; i < 4; ++i) {
      int e   = hf * 8 + i * 2;
      int off = hf * 16 + i * 2;
      f[e]     = p[off];
      f[e + 1] = p[off + 1];
    }
  return f;
}

// ---------------------------------------------------------------------------
// WMMA GEMM: Y = X[M,K]f16 @ W[N,K]f16^T + bias.
// One wave computes a 16(M) x (NT*16)(N) tile. NT/MODE/SILU are compile-time,
// so the K-loop is branch-free: NT b-frag loads + NT WMMAs per 32-K step,
// accumulators stay pinned (no exec masking around v_wmma).
// Requires N % (NT*16) == 0 (true for all call sites: 1024/NT=4, 16/NT=1).
// ---------------------------------------------------------------------------
template <int NT, int MODE, int SILU>
__global__ void gemm_wmma(const h16* __restrict__ X, const h16* __restrict__ W,
                          const float* __restrict__ bias, void* __restrict__ Y,
                          int M, int N, int K) {
  // scalarize wave index: uniform within wave -> SALU addressing + scalar branch
  int wave = __builtin_amdgcn_readfirstlane(
      blockIdx.x * (blockDim.x >> 5) + (threadIdx.x >> 5));
  int lane = threadIdx.x & 31;
  int mTiles  = (M + 15) >> 4;
  int nGroups = N / (NT * 16);
  int mt = wave % mTiles;
  int ng = wave / mTiles;
  if (ng >= nGroups) return;              // scalar guard (tail waves)
  int hi8 = (lane >= 16) ? 8 : 0;
  int l15 = lane & 15;
  int m0 = mt << 4;
  int arow = m0 + l15; if (arow >= M) arow = M - 1;   // clamp (store masked)

  v8f c[NT];
#pragma unroll
  for (int s = 0; s < NT; ++s) c[s] = v8f{};

  for (int k0 = 0; k0 < K; k0 += 32) {
    v16h a = load_frag(X, K, arow, k0, hi8);
#pragma unroll
    for (int s = 0; s < NT; ++s) {
      int n0 = (ng * NT + s) << 4;
      v16h b = load_frag(W, K, n0 + l15, k0, hi8);
      c[s] = __builtin_amdgcn_wmma_f32_16x16x32_f16(false, a, false, b,
                                                    (short)0, c[s], false, false);
    }
  }

#pragma unroll
  for (int s = 0; s < NT; ++s) {
    int n = ((ng * NT + s) << 4) + l15;
#pragma unroll
    for (int j = 0; j < 8; ++j) {
      int m = m0 + j + hi8;                // C layout: VGPR j -> M=j (+8 hi lanes)
      if (m >= M) continue;
      float v = c[s][j] + bias[n];
      if (SILU) v = v / (1.0f + __expf(-v));
      if (MODE == OUT_F32_PLAIN) {
        ((float*)Y)[(size_t)m * N + n] = v;
      } else if (MODE == OUT_F16_PLAIN) {
        ((h16*)Y)[(size_t)m * N + n] = (h16)v;
      } else if (MODE == OUT_F16_QK) {
        int b_ = m >> 11, s_ = m & (SS - 1);
        int h_ = n >> 6,  d_ = n & (DH - 1);
        ((h16*)Y)[(((size_t)b_ * NH + h_) * SS + s_) * DH + d_] = (h16)v;
      } else {  // OUT_F16_VT
        int b_ = m >> 11, s_ = m & (SS - 1);
        int h_ = n >> 6,  d_ = n & (DH - 1);
        ((h16*)Y)[(((size_t)b_ * NH + h_) * DH + d_) * SS + s_] = (h16)v;
      }
    }
  }
}

// f32 -> f16 conversion
__global__ void cvt_f16(const float* __restrict__ x, h16* __restrict__ y, int n) {
  int i = blockIdx.x * blockDim.x + threadIdx.x;
  if (i < n) y[i] = (h16)x[i];
}

// Position-bias MLP layer 0: h0[p,j] = silu((p-(S-1))*pw0[j] + pb0[j]), f16 out
__global__ void pos_mlp0(const float* __restrict__ pw0, const float* __restrict__ pb0,
                         h16* __restrict__ h0) {
  int i = blockIdx.x * blockDim.x + threadIdx.x;
  int p = i >> 10, j = i & 1023;
  if (p >= 2 * SS - 1) return;
  float x = (float)(p - (SS - 1));
  float v = x * pw0[j] + pb0[j];
  v = v / (1.0f + __expf(-v));
  h0[i] = (h16)v;
}

// ---------------------------------------------------------------------------
// Fused attention: one workgroup = (b, h, 16 q-rows).
// Phase 1: 8 waves compute scores (QK^T*scale + relbias, causal -inf) into LDS.
// Phase 2: block softmax over full 2048-wide rows; write attn to global once.
// Phase 3: waves split K for P@V WMMA; ds_add_f32 reduce; store ctx (f16).
// ---------------------------------------------------------------------------
__global__ void attn_kernel(const h16* __restrict__ q16, const h16* __restrict__ k16,
                            const h16* __restrict__ vT16, const float* __restrict__ hrel,
                            float* __restrict__ attn, h16* __restrict__ ctx16) {
  extern __shared__ float sc[];           // 16 * SS floats (128 KB dynamic)
  __shared__ float red[256];              // 16 rows x 16 partials
  __shared__ float ctxAcc[16 * DH];       // 16 x 64 f32

  int blk = blockIdx.x;
  int qt = blk & (SS / 16 - 1);
  int h  = (blk >> 7) & (NH - 1);
  int b  = blk >> 11;
  int q0 = qt << 4;

  int tid  = threadIdx.x;
  int wid  = __builtin_amdgcn_readfirstlane(tid >> 5);   // wave-uniform
  int lane = tid & 31;
  int hi8  = (lane >= 16) ? 8 : 0;
  int l15  = lane & 15;

  for (int i = tid; i < 16 * DH; i += blockDim.x) ctxAcc[i] = 0.0f;

  const h16* qbase = q16 + ((size_t)b * NH + h) * SS * DH;
  const h16* kbase = k16 + ((size_t)b * NH + h) * SS * DH;

  // Q fragments for this 16-row tile (K-dim = DH = 64 -> 2 frags)
  v16h a0 = load_frag(qbase, DH, q0 + l15, 0,  hi8);
  v16h a1 = load_frag(qbase, DH, q0 + l15, 32, hi8);

  // ---- Phase 1: scores -> LDS ----
#pragma unroll 1
  for (int i = 0; i < (SS / 16) / 8; ++i) {     // 16 col-tiles per wave
    int c0 = (i * 8 + wid) << 4;
    v16h b0 = load_frag(kbase, DH, c0 + l15, 0,  hi8);
    v16h b1 = load_frag(kbase, DH, c0 + l15, 32, hi8);
    v8f c = {};
    c = __builtin_amdgcn_wmma_f32_16x16x32_f16(false, a0, false, b0, (short)0, c, false, false);
    c = __builtin_amdgcn_wmma_f32_16x16x32_f16(false, a1, false, b1, (short)0, c, false, false);
#pragma unroll
    for (int j = 0; j < 8; ++j) {
      int qg = q0 + j + hi8;
      int kg = c0 + l15;
      float v = c[j] * SCALE + hrel[(size_t)(qg - kg + SS - 1) * NH + h];
      if (kg > qg) v = NEG_INF;           // causal mask
      sc[(j + hi8) * SS + kg] = v;
    }
  }
  __syncthreads();

  // ---- Phase 2: softmax (16 threads per row) ----
  int r = tid >> 4, jj = tid & 15;
  float mx = NEG_INF;
  for (int c = jj; c < SS; c += 16) mx = fmaxf(mx, sc[r * SS + c]);
  red[r * 16 + jj] = mx;
  __syncthreads();
  mx = NEG_INF;
#pragma unroll
  for (int t = 0; t < 16; ++t) mx = fmaxf(mx, red[r * 16 + t]);
  __syncthreads();
  float sum = 0.0f;
  for (int c = jj; c < SS; c += 16) {
    float e = __expf(sc[r * SS + c] - mx);   // exp(-inf)=0 for masked
    sc[r * SS + c] = e;
    sum += e;
  }
  red[r * 16 + jj] = sum;
  __syncthreads();
  sum = 0.0f;
#pragma unroll
  for (int t = 0; t < 16; ++t) sum += red[r * 16 + t];
  float inv = 1.0f / sum;
  size_t abase = (((size_t)b * NH + h) * SS + (q0 + r)) * SS;
  for (int c = jj; c < SS; c += 16) {
    float p = sc[r * SS + c] * inv;
    sc[r * SS + c] = p;
    attn[abase + c] = p;                  // the 537 MB output, written once
  }
  __syncthreads();

  // ---- Phase 3: P @ V, K split across 8 waves (256 k each) ----
  const h16* vb = vT16 + ((size_t)b * NH + h) * DH * SS;
  v8f cc[4];
#pragma unroll
  for (int s = 0; s < 4; ++s) cc[s] = v8f{};
  int kwBase = wid * (SS / 8);
#pragma unroll 1
  for (int kk = 0; kk < SS / 8; kk += 32) {
    int k0 = kwBase + kk;
    v16h a;                               // P fragment: LDS f32 -> f16
    const float* pr = &sc[l15 * SS + k0 + hi8];
#pragma unroll
    for (int hf = 0; hf < 2; ++hf)
#pragma unroll
      for (int ii = 0; ii < 4; ++ii) {
        int e   = hf * 8 + ii * 2;
        int off = hf * 16 + ii * 2;
        a[e]     = (h16)pr[off];
        a[e + 1] = (h16)pr[off + 1];
      }
#pragma unroll
    for (int s = 0; s < 4; ++s) {         // 4 n-tiles cover DH=64
      v16h bf = load_frag(vb, SS, s * 16 + l15, k0, hi8);
      cc[s] = __builtin_amdgcn_wmma_f32_16x16x32_f16(false, a, false, bf,
                                                     (short)0, cc[s], false, false);
    }
  }
#pragma unroll
  for (int s = 0; s < 4; ++s)
#pragma unroll
    for (int j = 0; j < 8; ++j)
      atomicAdd(&ctxAcc[(j + hi8) * DH + s * 16 + l15], cc[s][j]);  // ds_add_f32
  __syncthreads();

  for (int i = tid; i < 16 * DH; i += blockDim.x) {
    int m = i >> 6, d = i & (DH - 1);
    ctx16[((size_t)b * SS + q0 + m) * D_MODEL + h * DH + d] = (h16)ctxAcc[i];
  }
}

// ---------------------------------------------------------------------------
extern "C" void kernel_launch(void* const* d_in, const int* in_sizes, int n_in,
                              void* d_out, int out_size, void* d_ws, size_t ws_size,
                              hipStream_t stream) {
  (void)in_sizes; (void)n_in; (void)out_size; (void)ws_size;
  const float* query = (const float*)d_in[0];
  const float* key_  = (const float*)d_in[1];
  const float* value = (const float*)d_in[2];
  const float* wq = (const float*)d_in[3];  const float* bq = (const float*)d_in[4];
  const float* wk = (const float*)d_in[5];  const float* bk = (const float*)d_in[6];
  const float* wvp= (const float*)d_in[7];  const float* bv = (const float*)d_in[8];
  const float* wo = (const float*)d_in[9];  const float* bo = (const float*)d_in[10];
  const float* pw0= (const float*)d_in[11]; const float* pb0= (const float*)d_in[12];
  const float* pw1= (const float*)d_in[13]; const float* pb1= (const float*)d_in[14];
  const float* pw2= (const float*)d_in[15]; const float* pb2= (const float*)d_in[16];
  const float* pw3= (const float*)d_in[17]; const float* pb3= (const float*)d_in[18];

  // ---- workspace carve-out (~97 MB) ----
  char* ws = (char*)d_ws;
  size_t off = 0;
  auto alloc = [&](size_t bytes) -> void* {
    void* p = ws + off; off += (bytes + 255) & ~(size_t)255; return p;
  };
  const size_t M4 = (size_t)BB * SS;      // 4096 rows
  const size_t P  = 2 * SS - 1;           // 4095 relative positions

  h16* xq16 = (h16*)alloc(M4 * D_IN * 2);
  h16* xk16 = (h16*)alloc(M4 * D_IN * 2);
  h16* xv16 = (h16*)alloc(M4 * D_IN * 2);
  h16* wq16 = (h16*)alloc((size_t)D_MODEL * D_IN * 2);
  h16* wk16 = (h16*)alloc((size_t)D_MODEL * D_IN * 2);
  h16* wv16 = (h16*)alloc((size_t)D_MODEL * D_IN * 2);
  h16* wo16 = (h16*)alloc((size_t)D_MODEL * D_MODEL * 2);
  h16* pw116 = (h16*)alloc((size_t)D_IN * D_IN * 2);
  h16* pw216 = (h16*)alloc((size_t)D_IN * D_IN * 2);
  h16* pw316 = (h16*)alloc((size_t)NH * D_IN * 2);
  h16* h0 = (h16*)alloc(P * D_IN * 2);
  h16* h1 = (h16*)alloc(P * D_IN * 2);
  h16* h2 = (h16*)alloc(P * D_IN * 2);
  float* hrel = (float*)alloc(P * NH * 4);
  h16* q16  = (h16*)alloc((size_t)BB * NH * SS * DH * 2);
  h16* k16  = (h16*)alloc((size_t)BB * NH * SS * DH * 2);
  h16* vT16 = (h16*)alloc((size_t)BB * NH * DH * SS * 2);
  h16* ctx16 = (h16*)alloc(M4 * D_MODEL * 2);

  float* out  = (float*)d_out;
  float* attn = out + M4 * D_MODEL;       // tuple output #2

  auto cvt = [&](const float* s, h16* d, int n) {
    cvt_f16<<<(n + 255) / 256, 256, 0, stream>>>(s, d, n);
  };
  auto blocksFor = [](int M, int N, int NT) {
    int mTiles = (M + 15) / 16, nGroups = N / (NT * 16);
    return (mTiles * nGroups + 7) / 8;
  };

  // Convert activations + weights to f16
  cvt(query, xq16, (int)(M4 * D_IN));
  cvt(key_,  xk16, (int)(M4 * D_IN));
  cvt(value, xv16, (int)(M4 * D_IN));
  cvt(wq, wq16, D_MODEL * D_IN);
  cvt(wk, wk16, D_MODEL * D_IN);
  cvt(wvp, wv16, D_MODEL * D_IN);
  cvt(wo, wo16, D_MODEL * D_MODEL);
  cvt(pw1, pw116, D_IN * D_IN);
  cvt(pw2, pw216, D_IN * D_IN);
  cvt(pw3, pw316, NH * D_IN);

  // Relative-position-bias MLP -> hrel[2S-1, H]
  pos_mlp0<<<(int)((P * D_IN + 255) / 256), 256, 0, stream>>>(pw0, pb0, h0);
  gemm_wmma<4, OUT_F16_PLAIN, 1><<<blocksFor((int)P, D_IN, 4), 256, 0, stream>>>(
      h0, pw116, pb1, h1, (int)P, D_IN, D_IN);
  gemm_wmma<4, OUT_F16_PLAIN, 1><<<blocksFor((int)P, D_IN, 4), 256, 0, stream>>>(
      h1, pw216, pb2, h2, (int)P, D_IN, D_IN);
  gemm_wmma<1, OUT_F32_PLAIN, 0><<<blocksFor((int)P, NH, 1), 256, 0, stream>>>(
      h2, pw316, pb3, hrel, (int)P, NH, D_IN);

  // QKV projections into head-major layouts
  gemm_wmma<4, OUT_F16_QK, 0><<<blocksFor((int)M4, D_MODEL, 4), 256, 0, stream>>>(
      xq16, wq16, bq, q16, (int)M4, D_MODEL, D_IN);
  gemm_wmma<4, OUT_F16_QK, 0><<<blocksFor((int)M4, D_MODEL, 4), 256, 0, stream>>>(
      xk16, wk16, bk, k16, (int)M4, D_MODEL, D_IN);
  gemm_wmma<4, OUT_F16_VT, 0><<<blocksFor((int)M4, D_MODEL, 4), 256, 0, stream>>>(
      xv16, wv16, bv, vT16, (int)M4, D_MODEL, D_IN);

  // Fused attention (softmax + attn output + P@V), 128 KB dynamic LDS
  (void)hipFuncSetAttribute((const void*)attn_kernel,
                            hipFuncAttributeMaxDynamicSharedMemorySize, 16 * SS * 4);
  attn_kernel<<<BB * NH * (SS / 16), 256, 16 * SS * 4, stream>>>(
      q16, k16, vT16, hrel, attn, ctx16);

  // Output projection -> d_out (f32)
  gemm_wmma<4, OUT_F32_PLAIN, 0><<<blocksFor((int)M4, D_MODEL, 4), 256, 0, stream>>>(
      ctx16, wo16, bo, out, (int)M4, D_MODEL, D_MODEL);
}
